// DeepStreamOutput_29119878267614
// MI455X (gfx1250) — matrix operand order repaired and checked by reading, not compiled, not run on hardware
//
#include <hip/hip_runtime.h>
#include <hip/hip_bf16.h>
#include <math.h>

// ---------------------------------------------------------------------------
// YOLO-seg postprocess: best-class -> top400 -> NMS -> top100 -> WMMA GEMM
// (mask coeffs x protos) -> bilinear ROI sample + sigmoid.
// GEMM uses CDNA5 V_WMMA_F32_16X16X4_F32 (fp32-exact) when the builtin is
// declared, else the codegen-confirmed V_WMMA_F32_16X16X32_F16 path.
// ---------------------------------------------------------------------------

#define NCLS   80
#define NMSK   32
#define NANCH  8400
#define KTOP   400
#define MAXDET 100
#define CONF_T 0.25f
#define IOU_T  0.45f
#define CLSOFF 4096.0f
#define RSCALE 0.25f
#define PH     160
#define PW     160
#define NPIX   (PH * PW)          /* 25600 */
#define MPAD   112                /* MAXDET padded to multiple of 16 */
#define OUTC   (4 + 1 + 1 + NPIX) /* 25606 */
#define NTILE  (NPIX / 16)        /* 1600 */
#define MTILE  (MPAD / 16)        /* 7 */
#define NWAVES (NTILE * MTILE)    /* 11200 */

typedef float    v2f  __attribute__((ext_vector_type(2)));
typedef float    v8f  __attribute__((ext_vector_type(8)));
typedef _Float16 v16h __attribute__((ext_vector_type(16)));

// ---------------------------------------------------------------------------
// Kernel 1: per-anchor best class score + class id.
// preds layout: (116, 8400) channel-major; scores at channels [4, 84).
// ---------------------------------------------------------------------------
__global__ void k_best_cls(const float* __restrict__ preds,
                           float* __restrict__ bestv, int* __restrict__ bestc) {
  int n = blockIdx.x * blockDim.x + threadIdx.x;
  if (n >= NANCH) return;
  float bv = preds[4 * NANCH + n];
  int bc = 0;
  for (int c = 1; c < NCLS; ++c) {
    float v = preds[(4 + c) * NANCH + n];
    if (v > bv) { bv = v; bc = c; }  // strict > == argmax first-index tiebreak
  }
  bestv[n] = bv;
  bestc[n] = bc;
}

// ---------------------------------------------------------------------------
// Kernel 2: top-400 by iterative block-wide argmax (tie -> lower index),
// matching jax.lax.top_k ordering. Destroys bestv (scratch).
// ---------------------------------------------------------------------------
__global__ void k_top400(float* __restrict__ bestv, const int* __restrict__ bestc,
                         float* __restrict__ topv, int* __restrict__ topi) {
  __shared__ float sv[256];
  __shared__ int   si[256];
  const int tid = threadIdx.x;
  for (int t = 0; t < KTOP; ++t) {
    float bv = -__builtin_inff();
    int   bi = 0x7fffffff;
    for (int n = tid; n < NANCH; n += 256) {
      float v = bestv[n];
      if (v > bv || (v == bv && n < bi)) { bv = v; bi = n; }
    }
    sv[tid] = bv; si[tid] = bi;
    __syncthreads();
    for (int off = 128; off > 0; off >>= 1) {
      if (tid < off) {
        float v = sv[tid + off]; int i = si[tid + off];
        if (v > sv[tid] || (v == sv[tid] && i < si[tid])) { sv[tid] = v; si[tid] = i; }
      }
      __syncthreads();
    }
    if (tid == 0) {
      topv[t] = sv[0];
      topi[t] = si[0];
      bestv[si[0]] = -__builtin_inff();  // suppress for next round
    }
    __syncthreads();
  }
}

// ---------------------------------------------------------------------------
// Kernel 3: gather top-400 boxes/classes, greedy class-offset NMS.
// keep[j] &= !(keep[i] & iou(i,j)>T & j>i), i ascending (value-descending).
// ---------------------------------------------------------------------------
__global__ void k_nms(const float* __restrict__ preds,
                      const float* __restrict__ topv, const int* __restrict__ topi,
                      const int* __restrict__ bestc,
                      float* __restrict__ topb, int* __restrict__ topc,
                      int* __restrict__ keepw) {
  __shared__ float bx1[KTOP], by1[KTOP], bx2[KTOP], by2[KTOP], barea[KTOP];
  __shared__ int   keep[KTOP];
  const int tid = threadIdx.x;

  if (tid < KTOP) {
    int a = topi[tid];
    float x1 = preds[0 * NANCH + a];
    float y1 = preds[1 * NANCH + a];
    float x2 = preds[2 * NANCH + a];
    float y2 = preds[3 * NANCH + a];
    int   c  = bestc[a];
    topb[tid * 4 + 0] = x1; topb[tid * 4 + 1] = y1;
    topb[tid * 4 + 2] = x2; topb[tid * 4 + 3] = y2;
    topc[tid] = c;
    float off = (float)c * CLSOFF;  // class-offset trick
    bx1[tid] = x1 + off; by1[tid] = y1 + off;
    bx2[tid] = x2 + off; by2[tid] = y2 + off;
    float w = x2 - x1, h = y2 - y1;
    barea[tid] = fmaxf(w, 0.0f) * fmaxf(h, 0.0f);
    keep[tid] = (topv[tid] > CONF_T) ? 1 : 0;
  }
  for (int i = 0; i < KTOP; ++i) {
    __syncthreads();
    int ki = keep[i];  // stable during this iteration: writers have j > i
    if (ki && tid > i && tid < KTOP && keep[tid]) {
      float ix1 = fmaxf(bx1[i], bx1[tid]);
      float iy1 = fmaxf(by1[i], by1[tid]);
      float ix2 = fminf(bx2[i], bx2[tid]);
      float iy2 = fminf(by2[i], by2[tid]);
      float inter = fmaxf(ix2 - ix1, 0.0f) * fmaxf(iy2 - iy1, 0.0f);
      float iou = inter / (barea[i] + barea[tid] - inter + 1e-9f);
      if (iou > IOU_T) keep[tid] = 0;
    }
  }
  __syncthreads();
  if (tid < KTOP) keepw[tid] = keep[tid];
}

// ---------------------------------------------------------------------------
// Kernel 4: compact first <=100 kept entries (topv already descending, so
// ascending kept positions == jax top_k result). Writes output columns 0..5,
// sampling boxes (rbox), and the padded 112x32 coefficient matrix A.
// Invalid rows: cols 0..5 = 0, but mask branch uses anchor 0 (ref semantics).
// ---------------------------------------------------------------------------
__global__ void k_select(const float* __restrict__ preds,
                         const float* __restrict__ topv, const int* __restrict__ topi,
                         const float* __restrict__ topb, const int* __restrict__ topc,
                         const int* __restrict__ keepw,
                         float* __restrict__ out, float* __restrict__ rbox,
                         float* __restrict__ Amat) {
  __shared__ int sel[MAXDET];
  __shared__ int scnt;
  __shared__ int sridx[MAXDET];
  const int tid = threadIdx.x;
  if (tid == 0) {
    int cnt = 0;
    for (int j = 0; j < KTOP && cnt < MAXDET; ++j)
      if (keepw[j]) sel[cnt++] = j;
    scnt = cnt;
  }
  __syncthreads();
  const int cnt = scnt;
  for (int s = tid; s < MAXDET; s += blockDim.x) {
    if (s < cnt) {
      int j = sel[s];
      int a = topi[j];
      sridx[s] = a;
      float x1 = topb[j * 4 + 0], y1 = topb[j * 4 + 1];
      float x2 = topb[j * 4 + 2], y2 = topb[j * 4 + 3];
      out[s * OUTC + 0] = x1; out[s * OUTC + 1] = y1;
      out[s * OUTC + 2] = x2; out[s * OUTC + 3] = y2;
      out[s * OUTC + 4] = topv[j];
      out[s * OUTC + 5] = (float)topc[j];
      rbox[s * 4 + 0] = x1; rbox[s * 4 + 1] = y1;
      rbox[s * 4 + 2] = x2; rbox[s * 4 + 3] = y2;
    } else {
      sridx[s] = 0;  // reference: didx = 0 when invalid
      out[s * OUTC + 0] = 0.0f; out[s * OUTC + 1] = 0.0f;
      out[s * OUTC + 2] = 0.0f; out[s * OUTC + 3] = 0.0f;
      out[s * OUTC + 4] = 0.0f; out[s * OUTC + 5] = 0.0f;
      rbox[s * 4 + 0] = preds[0 * NANCH];  // boxes[anchor 0]
      rbox[s * 4 + 1] = preds[1 * NANCH];
      rbox[s * 4 + 2] = preds[2 * NANCH];
      rbox[s * 4 + 3] = preds[3 * NANCH];
    }
  }
  __syncthreads();
  for (int idx = tid; idx < MPAD * NMSK; idx += blockDim.x) {
    int s = idx >> 5, p = idx & 31;
    Amat[s * NMSK + p] = (s < MAXDET) ? preds[(84 + p) * NANCH + sridx[s]] : 0.0f;
  }
}

// ---------------------------------------------------------------------------
// Kernel 5: WMMA GEMM  combined(112x25600) = A(112x32) x protos(32x25600).
// One wave per 16x16 output tile; 11200 waves exactly (no divergence before
// WMMA -> EXEC all ones as the ISA requires).
// ---------------------------------------------------------------------------
__global__ void __launch_bounds__(256)
k_gemm_wmma(const float* __restrict__ Amat, const float* __restrict__ protos,
            float* __restrict__ comb) {
  const int wave  = blockIdx.x * (blockDim.x >> 5) + (threadIdx.x >> 5);
  const int lane  = threadIdx.x & 31;
  const int hi    = lane >> 4;   // lane half
  const int lm    = lane & 15;
  const int mtile = wave % MTILE;
  const int ntile = wave / MTILE;
  const int m0 = mtile * 16;
  const int col = ntile * 16 + lm;

  v8f acc = {0.f, 0.f, 0.f, 0.f, 0.f, 0.f, 0.f, 0.f};
  const float* Arow = Amat + (m0 + lm) * NMSK;

#if __has_builtin(__builtin_amdgcn_wmma_f32_16x16x4_f32)
  // fp32-exact path: 8 x V_WMMA_F32_16X16X4_F32 over K=32.
  // A 16x4 frag: VGPR0 = A[row][kb+2*hi], VGPR1 = A[row][kb+2*hi+1].
  // B 4x16 frag: VGPR0 = B[kb+2*hi][col], VGPR1 = B[kb+2*hi+1][col].
#pragma unroll
  for (int kk = 0; kk < 8; ++kk) {
    const int kb = kk * 4 + 2 * hi;
    v2f a, b;
    a.x = Arow[kb];
    a.y = Arow[kb + 1];
    b.x = protos[(size_t)kb * NPIX + col];
    b.y = protos[(size_t)(kb + 1) * NPIX + col];
    acc = __builtin_amdgcn_wmma_f32_16x16x4_f32(false, a, false, b,
                                                (short)0, acc, false, false);
  }
#else
  // Fallback: codegen-confirmed V_WMMA_F32_16X16X32_F16, one op per tile.
  // 16-bit A 16x32: VGPR v(0..3): K = 8*hi + 2v + s ; v(4..7): K = 16 + 8*hi + 2(v-4) + s.
  // 16-bit B 32x16: VGPR v(0..3): K = 8*hi + 2v + s ; v(4..7): K = 16 + 8*hi + 2(v-4) + s, N = lm.
  v16h af, bf;
#pragma unroll
  for (int v = 0; v < 8; ++v) {
#pragma unroll
    for (int s = 0; s < 2; ++s) {
      int kA = (v < 4) ? (8 * hi + 2 * v + s) : (16 + 8 * hi + 2 * (v - 4) + s);
      int kB = kA;
      af[2 * v + s] = (_Float16)Arow[kA];
      bf[2 * v + s] = (_Float16)protos[(size_t)kB * NPIX + col];
    }
  }
  acc = __builtin_amdgcn_wmma_f32_16x16x32_f16(false, af, false, bf,
                                               (short)0, acc, false, false);
#endif

  // D layout: VGPR v -> row M = v + 8*hi, N = lm.
#pragma unroll
  for (int v = 0; v < 8; ++v) {
    int row = m0 + v + 8 * hi;
    if (row < MAXDET) comb[(size_t)row * NPIX + col] = acc[v];
  }
}

// ---------------------------------------------------------------------------
// Kernel 6: bilinear-sample combined[r] at the ROI grid + sigmoid.
// Matches reference _roi_align index clamping / fractional weights exactly.
// ---------------------------------------------------------------------------
__global__ void k_sample_sigmoid(const float* __restrict__ comb,
                                 const float* __restrict__ rbox,
                                 float* __restrict__ out) {
  const int r   = blockIdx.y;
  const int pix = blockIdx.x * blockDim.x + threadIdx.x;
  if (pix >= NPIX) return;
  const int oy = pix / PW, ox = pix % PW;

  const float x1 = rbox[r * 4 + 0] * RSCALE - 0.5f;
  const float y1 = rbox[r * 4 + 1] * RSCALE - 0.5f;
  const float x2 = rbox[r * 4 + 2] * RSCALE - 0.5f;
  const float y2 = rbox[r * 4 + 3] * RSCALE - 0.5f;
  const float bw = (x2 - x1) / (float)PW;
  const float bh = (y2 - y1) / (float)PH;
  const float sx = x1 + ((float)ox + 0.5f) * bw;
  const float sy = y1 + ((float)oy + 0.5f) * bh;

  const float x0f = floorf(sx), y0f = floorf(sy);
  const float lx = sx - x0f, ly = sy - y0f;
  int x0 = min(max((int)x0f, 0), PW - 1);
  int x1i = min(max((int)x0f + 1, 0), PW - 1);
  int y0 = min(max((int)y0f, 0), PH - 1);
  int y1i = min(max((int)y0f + 1, 0), PH - 1);

  const float* cr = comb + (size_t)r * NPIX;
  float v00 = cr[y0 * PW + x0];
  float v01 = cr[y0 * PW + x1i];
  float v10 = cr[y1i * PW + x0];
  float v11 = cr[y1i * PW + x1i];
  float val = v00 * (1.f - ly) * (1.f - lx) + v01 * (1.f - ly) * lx +
              v10 * ly * (1.f - lx) + v11 * ly * lx;
  out[(size_t)r * OUTC + 6 + pix] = 1.0f / (1.0f + expf(-val));
}

// ---------------------------------------------------------------------------
extern "C" void kernel_launch(void* const* d_in, const int* in_sizes, int n_in,
                              void* d_out, int out_size, void* d_ws, size_t ws_size,
                              hipStream_t stream) {
  const float* preds  = (const float*)d_in[0];   // (1, 116, 8400) f32
  const float* protos = (const float*)d_in[1];   // (1, 32, 160, 160) f32
  float* out = (float*)d_out;                    // (1, 100, 25606) f32
  float* ws  = (float*)d_ws;

  // Workspace layout (float units)
  float* bestv = ws;                        // 8400
  int*   bestc = (int*)(ws + 8400);         // 8400
  float* topv  = ws + 16800;                // 400
  int*   topi  = (int*)(ws + 17200);        // 400
  float* topb  = ws + 17600;                // 1600
  int*   topc  = (int*)(ws + 19200);        // 400
  int*   keepw = (int*)(ws + 19600);        // 400
  float* rbox  = ws + 20000;                // 400
  float* Amat  = ws + 20400;                // 112*32 = 3584
  float* comb  = ws + 24000;                // 100*25600 = 2,560,000

  k_best_cls<<<(NANCH + 255) / 256, 256, 0, stream>>>(preds, bestv, bestc);
  k_top400<<<1, 256, 0, stream>>>(bestv, bestc, topv, topi);
  k_nms<<<1, 512, 0, stream>>>(preds, topv, topi, bestc, topb, topc, keepw);
  k_select<<<1, 128, 0, stream>>>(preds, topv, topi, topb, topc, keepw,
                                  out, rbox, Amat);
  k_gemm_wmma<<<NWAVES / 8, 256, 0, stream>>>(Amat, protos, comb);
  k_sample_sigmoid<<<dim3(NPIX / 256, MAXDET), 256, 0, stream>>>(comb, rbox, out);
}
